// StockAutoformer_28467043238448
// MI455X (gfx1250) — compile-verified, matrix operand order checked by reading, not run on hardware
//
#include <hip/hip_runtime.h>
#include <hip/hip_bf16.h>
#include <math.h>

// ---------------------------------------------------------------------------
// Model dims (StockAutoformer reference)
// ---------------------------------------------------------------------------
#define BB   64      // batch
#define TT   512     // encoder sequence length
#define CC   8       // input channels
#define DD   512     // model dim
#define HH   8       // heads
#define DH   64      // head dim
#define FF   2048    // ff hidden
#define TDQ  120     // decoder seq (LABEL 96 + PRED 24)
#define PREDN 24
#define KK_ENC 6     // int(log(512))
#define KK_DEC 4     // int(log(120))

// GEMM tiling: block tile 128x64, BK=32; 8 waves, each wave a 32x32 register
// tile (2x2 WMMA subtiles -> 4 v_wmma per K-step).
#define BM 128
#define BN 64
#define BK 32
#define BKP 40       // padded LDS row stride in ushorts (80B: 16B-aligned, bank-spread)

// ---------------------------------------------------------------------------
// bf16 WMMA plumbing
// ---------------------------------------------------------------------------
typedef __bf16 bf16_t;
typedef bf16_t v16bf __attribute__((ext_vector_type(16)));
typedef float  v8f   __attribute__((ext_vector_type(8)));

union BF16Frag { v16bf v; unsigned short u[16]; uint4 q[2]; };

__device__ __forceinline__ unsigned short f32_to_bf16(float f) {
  union { float f; unsigned u; } cv; cv.f = f;
  unsigned u = cv.u;
  u += 0x7FFFu + ((u >> 16) & 1u);   // round-to-nearest-even
  return (unsigned short)(u >> 16);
}
__device__ __forceinline__ unsigned pack_bf16x2(float lo, float hi) {
  return (unsigned)f32_to_bf16(lo) | ((unsigned)f32_to_bf16(hi) << 16);
}

// ---------------------------------------------------------------------------
// GEMM: C[M,N] = A[M,K] @ W[K,N] + bias[N]   (f32 in/out, bf16 WMMA core)
//
// Wave w: wm = (w>>1)*32, wn = (w&1)*32; subtiles rows wm+{0,16}, cols
// wn+{0,16}. A staged [BM][BK] bf16 row-major; B staged transposed [BN][BK];
// both rows padded to 80B (16B aligned, 16 distinct banks across l16).
// Fragments: two 16B LDS reads each; fast paths fully uniform.
//
// A-fragment (16-bit A 16x32, ISA 7.12.2): lane<16: M=lane, elems j -> K =
// 8*half + j (j<8) / 16+8*half+(j-8). B mirrored with col = lane&15.
// C/D: VGPR r -> M = r + 8*half, N = lane&15.
// ---------------------------------------------------------------------------
__global__ __launch_bounds__(256)
void k_gemm_wmma_bf16(const float* __restrict__ A, const float* __restrict__ W,
                      const float* __restrict__ bias, float* __restrict__ C,
                      int M, int N, int K) {
  __shared__ alignas(16) unsigned short As[BM][BKP];
  __shared__ alignas(16) unsigned short Bs[BN][BKP];

  int tid  = threadIdx.x;
  int wave = tid >> 5, lane = tid & 31;
  int half = lane >> 4, l16 = lane & 15;

  int tilesN = (N + BN - 1) / BN;
  int bm = (blockIdx.x / tilesN) * BM;
  int bn = (blockIdx.x % tilesN) * BN;
  int wm = (wave >> 1) * 32;
  int wn = (wave & 1) * 32;

  // staging coordinates
  int ar  = tid >> 1;           // A row 0..127
  int ak0 = (tid & 1) * 16;     // A k start 0 or 16
  int bk  = tid >> 3;           // B k row 0..31
  int bn0 = (tid & 7) * 8;      // B n start 0..56

  v8f acc00 = {0.f,0.f,0.f,0.f,0.f,0.f,0.f,0.f};
  v8f acc01 = acc00, acc10 = acc00, acc11 = acc00;

  for (int kb = 0; kb < K; kb += BK) {
    // ---- stage A (128x32) ----
    if (bm + BM <= M && kb + BK <= K) {
      const float* src = A + (size_t)(bm + ar) * K + kb + ak0;
      float4 v0 = *(const float4*)(src);
      float4 v1 = *(const float4*)(src + 4);
      float4 v2 = *(const float4*)(src + 8);
      float4 v3 = *(const float4*)(src + 12);
      unsigned* dst = (unsigned*)&As[ar][ak0];
      dst[0] = pack_bf16x2(v0.x, v0.y);
      dst[1] = pack_bf16x2(v0.z, v0.w);
      dst[2] = pack_bf16x2(v1.x, v1.y);
      dst[3] = pack_bf16x2(v1.z, v1.w);
      dst[4] = pack_bf16x2(v2.x, v2.y);
      dst[5] = pack_bf16x2(v2.z, v2.w);
      dst[6] = pack_bf16x2(v3.x, v3.y);
      dst[7] = pack_bf16x2(v3.z, v3.w);
    } else {
      int grow = bm + ar;
#pragma unroll
      for (int j = 0; j < 16; ++j) {
        int gk = kb + ak0 + j;
        float v = (grow < M && gk < K) ? A[(size_t)grow * K + gk] : 0.f;
        As[ar][ak0 + j] = f32_to_bf16(v);
      }
    }
    // ---- stage B (32x64, stored transposed [n][k]) ----
    if (bn + BN <= N && kb + BK <= K) {
      const float* src = W + (size_t)(kb + bk) * N + bn + bn0;
      float4 v0 = *(const float4*)(src);
      float4 v1 = *(const float4*)(src + 4);
      Bs[bn0 + 0][bk] = f32_to_bf16(v0.x);
      Bs[bn0 + 1][bk] = f32_to_bf16(v0.y);
      Bs[bn0 + 2][bk] = f32_to_bf16(v0.z);
      Bs[bn0 + 3][bk] = f32_to_bf16(v0.w);
      Bs[bn0 + 4][bk] = f32_to_bf16(v1.x);
      Bs[bn0 + 5][bk] = f32_to_bf16(v1.y);
      Bs[bn0 + 6][bk] = f32_to_bf16(v1.z);
      Bs[bn0 + 7][bk] = f32_to_bf16(v1.w);
    } else {
      int gk = kb + bk;
#pragma unroll
      for (int j = 0; j < 8; ++j) {
        int gn = bn + bn0 + j;
        float v = (gk < K && gn < N) ? W[(size_t)gk * N + gn] : 0.f;
        Bs[bn0 + j][bk] = f32_to_bf16(v);
      }
    }
    // prefetch next K chunk (speculative; lowers to global_prefetch_b8)
    if (kb + BK < K) {
      __builtin_prefetch(A + (size_t)(bm + ar) * K + kb + BK + ak0, 0, 3);
      __builtin_prefetch(W + (size_t)(kb + BK + bk) * N + bn + bn0, 0, 3);
    }
    __syncthreads();

    // ---- fragments + 4x WMMA (branch-free) ----
    BF16Frag a0, a1, b0, b1;
    a0.q[0] = *(const uint4*)&As[wm + l16][8 * half];
    a0.q[1] = *(const uint4*)&As[wm + l16][16 + 8 * half];
    a1.q[0] = *(const uint4*)&As[wm + 16 + l16][8 * half];
    a1.q[1] = *(const uint4*)&As[wm + 16 + l16][16 + 8 * half];
    b0.q[0] = *(const uint4*)&Bs[wn + l16][8 * half];
    b0.q[1] = *(const uint4*)&Bs[wn + l16][16 + 8 * half];
    b1.q[0] = *(const uint4*)&Bs[wn + 16 + l16][8 * half];
    b1.q[1] = *(const uint4*)&Bs[wn + 16 + l16][16 + 8 * half];

    acc00 = __builtin_amdgcn_wmma_f32_16x16x32_bf16(
        false, a0.v, false, b0.v, (short)0, acc00, false, false);
    acc01 = __builtin_amdgcn_wmma_f32_16x16x32_bf16(
        false, a0.v, false, b1.v, (short)0, acc01, false, false);
    acc10 = __builtin_amdgcn_wmma_f32_16x16x32_bf16(
        false, a1.v, false, b0.v, (short)0, acc10, false, false);
    acc11 = __builtin_amdgcn_wmma_f32_16x16x32_bf16(
        false, a1.v, false, b1.v, (short)0, acc11, false, false);
    __syncthreads();
  }

  // ---- store (+bias) ----
  int n0 = bn + wn + l16;
  int n1 = n0 + 16;
  if (bm + BM <= M && bn + BN <= N) {
    float bi0 = bias[n0];
    float bi1 = bias[n1];
#pragma unroll
    for (int r = 0; r < 8; ++r) {
      int m0 = bm + wm + r + 8 * half;
      int m1 = m0 + 16;
      C[(size_t)m0 * N + n0] = acc00[r] + bi0;
      C[(size_t)m0 * N + n1] = acc01[r] + bi1;
      C[(size_t)m1 * N + n0] = acc10[r] + bi0;
      C[(size_t)m1 * N + n1] = acc11[r] + bi1;
    }
  } else {
    float bi0 = (n0 < N) ? bias[n0] : 0.f;
    float bi1 = (n1 < N) ? bias[n1] : 0.f;
#pragma unroll
    for (int r = 0; r < 8; ++r) {
      int m0 = bm + wm + r + 8 * half;
      int m1 = m0 + 16;
      if (m0 < M && n0 < N) C[(size_t)m0 * N + n0] = acc00[r] + bi0;
      if (m0 < M && n1 < N) C[(size_t)m0 * N + n1] = acc01[r] + bi1;
      if (m1 < M && n0 < N) C[(size_t)m1 * N + n0] = acc10[r] + bi0;
      if (m1 < M && n1 < N) C[(size_t)m1 * N + n1] = acc11[r] + bi1;
    }
  }
}

// ---------------------------------------------------------------------------
// y[row,:] = LayerNorm(res[row,:] + x[row,:]) * g + b      (res may be null)
// ---------------------------------------------------------------------------
__global__ __launch_bounds__(256)
void k_add_layernorm(const float* __restrict__ res, const float* __restrict__ x,
                     const float* __restrict__ g, const float* __restrict__ b,
                     float* __restrict__ y, int D) {
  int row = blockIdx.x;
  const float* xr = x + (size_t)row * D;
  const float* rr = res ? res + (size_t)row * D : nullptr;
  __shared__ float s1[256], s2[256];
  float sum = 0.f, sq = 0.f;
  for (int i = threadIdx.x; i < D; i += blockDim.x) {
    float v = xr[i] + (rr ? rr[i] : 0.f);
    sum += v; sq += v * v;
  }
  s1[threadIdx.x] = sum; s2[threadIdx.x] = sq;
  __syncthreads();
  for (int s = blockDim.x >> 1; s > 0; s >>= 1) {
    if (threadIdx.x < (unsigned)s) {
      s1[threadIdx.x] += s1[threadIdx.x + s];
      s2[threadIdx.x] += s2[threadIdx.x + s];
    }
    __syncthreads();
  }
  float mean = s1[0] / D;
  float var  = s2[0] / D - mean * mean;
  float rstd = rsqrtf(var + 1e-5f);
  for (int i = threadIdx.x; i < D; i += blockDim.x) {
    float v = xr[i] + (rr ? rr[i] : 0.f);
    y[(size_t)row * D + i] = (v - mean) * rstd * g[i] + b[i];
  }
}

// ---------------------------------------------------------------------------
// Seasonal part of series decomposition: y = x - movavg_25(edge-padded x)
// ---------------------------------------------------------------------------
__global__ __launch_bounds__(256)
void k_decomp_seasonal(const float* __restrict__ x, float* __restrict__ y,
                       int Bn, int T, int D) {
  size_t idx = (size_t)blockIdx.x * blockDim.x + threadIdx.x;
  size_t total = (size_t)Bn * T * D;
  if (idx >= total) return;
  int d = (int)(idx % D);
  size_t bt = idx / D;
  int t = (int)(bt % T);
  int b = (int)(bt / T);
  float s = 0.f;
#pragma unroll
  for (int j = -12; j <= 12; ++j) {
    int tt = t + j;
    tt = tt < 0 ? 0 : (tt >= T ? T - 1 : tt);
    s += x[((size_t)b * T + tt) * D + d];
  }
  y[idx] = x[idx] - s * (1.0f / 25.0f);
}

// ---------------------------------------------------------------------------
// corr_mean[b,h,tau] = (1/dh) * sum_{t<Tq} sum_d Q[b,t,h,d]*K[b,(t-tau)%Tk,h,d]
// (== mean over d of irfft(rfft(Q,n=Tk)*conj(rfft(K,n=Tk)))[:Tq])
// ---------------------------------------------------------------------------
__global__ __launch_bounds__(256)
void k_autocorr_mean(const float* __restrict__ Q, const float* __restrict__ Kv,
                     float* __restrict__ cm, int Bn, int H, int dh,
                     int Tq, int Tk, int D) {
  int idx = blockIdx.x * blockDim.x + threadIdx.x;
  if (idx >= Bn * H * Tq) return;
  int tau = idx % Tq;
  int bh  = idx / Tq;
  int h   = bh % H;
  int b   = bh / H;
  const float* Qb = Q  + (size_t)b * Tq * D + h * dh;
  const float* Kb = Kv + (size_t)b * Tk * D + h * dh;
  float s = 0.f;
  for (int t = 0; t < Tq; ++t) {
    int tk = t - tau; if (tk < 0) tk += Tk;
    const float* qr = Qb + (size_t)t  * D;
    const float* kr = Kb + (size_t)tk * D;
    float acc = 0.f;
    for (int d = 0; d < dh; ++d) acc += qr[d] * kr[d];
    s += acc;
  }
  cm[idx] = s / (float)dh;
}

// top-k (descending, first occurrence) per (b,h) over Tq lags
__global__ __launch_bounds__(256)
void k_topk(const float* __restrict__ cm, float* __restrict__ vals,
            int* __restrict__ lags, int BH, int Tq, int kk) {
  int bh = blockIdx.x * blockDim.x + threadIdx.x;
  if (bh >= BH) return;
  const float* c = cm + (size_t)bh * Tq;
  int chosen[8];
  for (int i = 0; i < kk; ++i) {
    float best = -3.4e38f; int bi = 0;
    for (int t = 0; t < Tq; ++t) {
      bool used = false;
      for (int j = 0; j < i; ++j) used = used || (chosen[j] == t);
      if (!used && c[t] > best) { best = c[t]; bi = t; }
    }
    chosen[i] = bi;
    vals[(size_t)bh * kk + i] = best;
    lags[(size_t)bh * kk + i] = bi;
  }
}

// lag_i = floor(mean over (b,h) of lags[:,:,i])   (tiny, single thread)
__global__ void k_lag_mean(const int* __restrict__ lags, int* __restrict__ lag_out,
                           int BH, int kk) {
  if (threadIdx.x != 0 || blockIdx.x != 0) return;
  for (int i = 0; i < kk; ++i) {
    float s = 0.f;
    for (int bh = 0; bh < BH; ++bh) s += (float)lags[bh * kk + i];
    lag_out[i] = (int)floorf(s / (float)BH);
  }
}

// in-place softmax over kk per (b,h)
__global__ __launch_bounds__(256)
void k_softmax_k(float* __restrict__ vals, int BH, int kk) {
  int bh = blockIdx.x * blockDim.x + threadIdx.x;
  if (bh >= BH) return;
  float* v = vals + (size_t)bh * kk;
  float m = v[0];
  for (int i = 1; i < kk; ++i) m = fmaxf(m, v[i]);
  float s = 0.f;
  for (int i = 0; i < kk; ++i) { float e = expf(v[i] - m); v[i] = e; s += e; }
  for (int i = 0; i < kk; ++i) v[i] /= s;
}

// out[b,t, h*dh+d] = sum_i w[b,h,i] * V[b,(t+lag_i)%Tk, h*dh+d]
__global__ __launch_bounds__(256)
void k_lag_aggregate(const float* __restrict__ V, const float* __restrict__ w,
                     const int* __restrict__ lag, float* __restrict__ out,
                     int Bn, int H, int dh, int Tq, int Tk, int kk, int D) {
  size_t idx = (size_t)blockIdx.x * blockDim.x + threadIdx.x;
  size_t total = (size_t)Bn * Tq * D;
  if (idx >= total) return;
  int c = (int)(idx % D);
  size_t bt = idx / D;
  int t = (int)(bt % Tq);
  int b = (int)(bt / Tq);
  int h = c / dh;
  float s = 0.f;
  for (int i = 0; i < kk; ++i) {
    int tt = t + lag[i]; if (tt >= Tk) tt -= Tk;
    s += w[((size_t)b * H + h) * kk + i] * V[((size_t)b * Tk + tt) * D + c];
  }
  out[idx] = s;
}

__global__ __launch_bounds__(256)
void k_gelu(float* __restrict__ x, size_t n) {
  size_t i = (size_t)blockIdx.x * blockDim.x + threadIdx.x;
  if (i < n) {
    float v = x[i];
    x[i] = 0.5f * v * (1.0f + erff(v * 0.70710678118654752f));
  }
}

// decoder input: [B,120,8] = concat(x[:, -96:, :], zeros(B,24,8))
__global__ __launch_bounds__(256)
void k_build_dec_input(const float* __restrict__ x, float* __restrict__ y) {
  int i = blockIdx.x * blockDim.x + threadIdx.x;
  int total = BB * TDQ * CC;
  if (i >= total) return;
  int c = i % CC;
  int t = (i / CC) % TDQ;
  int b = i / (CC * TDQ);
  y[i] = (t < TDQ - PREDN) ? x[((size_t)b * TT + (TT - (TDQ - PREDN) + t)) * CC + c]
                           : 0.f;
}

// pooled[b,d] = mean over last P rows of dec[b, :, d]
__global__ __launch_bounds__(256)
void k_pool_tail(const float* __restrict__ dec, float* __restrict__ pooled,
                 int Bn, int T, int D, int P) {
  int i = blockIdx.x * blockDim.x + threadIdx.x;
  if (i >= Bn * D) return;
  int d = i % D;
  int b = i / D;
  float s = 0.f;
  for (int t = T - P; t < T; ++t) s += dec[((size_t)b * T + t) * D + d];
  pooled[i] = s / (float)P;
}

__global__ __launch_bounds__(256)
void k_clip(float* __restrict__ x, int n, float lo, float hi) {
  int i = blockIdx.x * blockDim.x + threadIdx.x;
  if (i < n) x[i] = fminf(fmaxf(x[i], lo), hi);
}

// ---------------------------------------------------------------------------
// Host launcher — input order: d_in[0]=x, then params flattened with dicts in
// sorted-key order (jax pytree convention). See enum.
// ---------------------------------------------------------------------------
enum {
  IN_X = 0,
  D_CROSS_KB = 1, D_CROSS_KW, D_CROSS_OB, D_CROSS_OW,
  D_CROSS_QB, D_CROSS_QW, D_CROSS_VB, D_CROSS_VW,
  D_FF1B, D_FF1W, D_FF2B, D_FF2W,
  D_LN1B, D_LN1G, D_LN2B, D_LN2G, D_LN3B, D_LN3G,
  D_SELF_KB, D_SELF_KW, D_SELF_OB, D_SELF_OW,
  D_SELF_QB, D_SELF_QW, D_SELF_VB, D_SELF_VW,
  D_TRENDW,
  DEC_EMB_B, DEC_EMB_W, DEC_NORM_B, DEC_NORM_G,
  E0_ATT_KB, E0_ATT_KW, E0_ATT_OB, E0_ATT_OW,
  E0_ATT_QB, E0_ATT_QW, E0_ATT_VB, E0_ATT_VW,
  E0_FF1B, E0_FF1W, E0_FF2B, E0_FF2W,
  E0_LN1B, E0_LN1G, E0_LN2B, E0_LN2G,
  E1_ATT_KB, E1_ATT_KW, E1_ATT_OB, E1_ATT_OW,
  E1_ATT_QB, E1_ATT_QW, E1_ATT_VB, E1_ATT_VW,
  E1_FF1B, E1_FF1W, E1_FF2B, E1_FF2W,
  E1_LN1B, E1_LN1G, E1_LN2B, E1_LN2G,
  ENC_EMB_B, ENC_EMB_W, ENC_NORM_B, ENC_NORM_G,
  LV_B, LV_W, MEAN_B, MEAN_W
};

static inline int cdiv(long long a, long long b) { return (int)((a + b - 1) / b); }

extern "C" void kernel_launch(void* const* d_in, const int* in_sizes, int n_in,
                              void* d_out, int out_size, void* d_ws, size_t ws_size,
                              hipStream_t stream) {
  (void)in_sizes; (void)n_in; (void)out_size; (void)ws_size;
  const float* x = (const float*)d_in[IN_X];
  auto P = [&](int i) { return (const float*)d_in[i]; };

  // workspace carve-out (floats)
  size_t off = 0;
  auto alloc = [&](size_t n) { float* p = (float*)d_ws + off; off += n; return p; };
  const size_t NE = (size_t)BB * TT;    // 32768 encoder rows
  const size_t ND = (size_t)BB * TDQ;   //  7680 decoder rows
  float* bufX    = alloc(NE * DD);
  float* bufQ    = alloc(NE * DD);
  float* bufK    = alloc(NE * DD);
  float* bufV    = alloc(NE * DD);
  float* bufT    = alloc(NE * DD);
  float* encOut  = alloc(NE * DD);
  float* bufFF   = alloc(NE * FF);
  float* bufDec  = alloc(ND * DD);
  float* bufDecT = alloc(ND * DD);
  float* bufDecS = alloc(ND * DD);
  float* corr    = alloc((size_t)BB * HH * TT);
  float* vals    = alloc((size_t)BB * HH * 8);
  int*   lags    = (int*)alloc((size_t)BB * HH * 8);
  int*   lagm    = (int*)alloc(16);
  float* decinp  = alloc(ND * CC);
  float* pooled  = alloc((size_t)BB * DD);

  auto gemm = [&](const float* A, const float* W, const float* bias, float* Cst,
                  int M, int N, int K) {
    int blocks = cdiv(M, BM) * cdiv(N, BN);
    k_gemm_wmma_bf16<<<blocks, 256, 0, stream>>>(A, W, bias, Cst, M, N, K);
  };
  auto addln = [&](const float* res, const float* xin, const float* g,
                   const float* b, float* y, int rows) {
    k_add_layernorm<<<rows, 256, 0, stream>>>(res, xin, g, b, y, DD);
  };
  auto decomp = [&](const float* xin, float* y, int Bn, int T) {
    size_t tot = (size_t)Bn * T * DD;
    k_decomp_seasonal<<<cdiv((long long)tot, 256), 256, 0, stream>>>(xin, y, Bn, T, DD);
  };
  auto autocorr = [&](const float* qsrc, const float* kvsrc, int Tq, int Tk, int kk,
                      const float* wq, const float* bq, const float* wk, const float* bk,
                      const float* wv, const float* bv, const float* wo, const float* bo,
                      float* preO, float* out) {
    int Mq = BB * Tq, Mk = BB * Tk;
    gemm(qsrc,  wq, bq, bufQ, Mq, DD, DD);
    gemm(kvsrc, wk, bk, bufK, Mk, DD, DD);
    gemm(kvsrc, wv, bv, bufV, Mk, DD, DD);
    int ncorr = BB * HH * Tq;
    k_autocorr_mean<<<cdiv(ncorr, 256), 256, 0, stream>>>(bufQ, bufK, corr,
                                                          BB, HH, DH, Tq, Tk, DD);
    int BH = BB * HH;
    k_topk<<<cdiv(BH, 256), 256, 0, stream>>>(corr, vals, lags, BH, Tq, kk);
    k_lag_mean<<<1, 1, 0, stream>>>(lags, lagm, BH, kk);
    k_softmax_k<<<cdiv(BH, 256), 256, 0, stream>>>(vals, BH, kk);
    size_t nagg = (size_t)BB * Tq * DD;
    k_lag_aggregate<<<cdiv((long long)nagg, 256), 256, 0, stream>>>(
        bufV, vals, lagm, preO, BB, HH, DH, Tq, Tk, kk, DD);
    gemm(preO, wo, bo, out, Mq, DD, DD);
  };

  // ---- encoder ----
  gemm(x, P(ENC_EMB_W), P(ENC_EMB_B), bufX, (int)NE, DD, CC);
  const int encIdx[2][16] = {
    {E0_ATT_KB,E0_ATT_KW,E0_ATT_OB,E0_ATT_OW,E0_ATT_QB,E0_ATT_QW,E0_ATT_VB,E0_ATT_VW,
     E0_FF1B,E0_FF1W,E0_FF2B,E0_FF2W,E0_LN1B,E0_LN1G,E0_LN2B,E0_LN2G},
    {E1_ATT_KB,E1_ATT_KW,E1_ATT_OB,E1_ATT_OW,E1_ATT_QB,E1_ATT_QW,E1_ATT_VB,E1_ATT_VW,
     E1_FF1B,E1_FF1W,E1_FF2B,E1_FF2W,E1_LN1B,E1_LN1G,E1_LN2B,E1_LN2G}};
  for (int l = 0; l < 2; ++l) {
    const int* I = encIdx[l];
    autocorr(bufX, bufX, TT, TT, KK_ENC,
             P(I[5]), P(I[4]), P(I[1]), P(I[0]),
             P(I[7]), P(I[6]), P(I[3]), P(I[2]),
             bufT /*preO*/, bufQ /*attn out*/);
    addln(bufX, bufQ, P(I[13]), P(I[12]), bufT, (int)NE);   // ln1(res + attn)
    decomp(bufT, bufX, BB, TT);
    gemm(bufX, P(I[9]), P(I[8]), bufFF, (int)NE, FF, DD);
    k_gelu<<<cdiv((long long)(NE * FF), 256), 256, 0, stream>>>(bufFF, NE * FF);
    gemm(bufFF, P(I[11]), P(I[10]), bufQ, (int)NE, DD, FF);
    addln(bufX, bufQ, P(I[15]), P(I[14]), bufT, (int)NE);   // ln2(res + ff)
    decomp(bufT, bufX, BB, TT);
  }
  addln(nullptr, bufX, P(ENC_NORM_G), P(ENC_NORM_B), encOut, (int)NE);

  // ---- decoder ----
  k_build_dec_input<<<cdiv(BB * TDQ * CC, 256), 256, 0, stream>>>(x, decinp);
  gemm(decinp, P(DEC_EMB_W), P(DEC_EMB_B), bufDec, (int)ND, DD, CC);

  autocorr(bufDec, bufDec, TDQ, TDQ, KK_DEC,
           P(D_SELF_QW), P(D_SELF_QB), P(D_SELF_KW), P(D_SELF_KB),
           P(D_SELF_VW), P(D_SELF_VB), P(D_SELF_OW), P(D_SELF_OB),
           bufDecS, bufDecT);
  addln(bufDec, bufDecT, P(D_LN1G), P(D_LN1B), bufDecS, (int)ND);
  decomp(bufDecS, bufDec, BB, TDQ);

  autocorr(bufDec, encOut, TDQ, TT, KK_DEC,
           P(D_CROSS_QW), P(D_CROSS_QB), P(D_CROSS_KW), P(D_CROSS_KB),
           P(D_CROSS_VW), P(D_CROSS_VB), P(D_CROSS_OW), P(D_CROSS_OB),
           bufDecS, bufDecT);
  addln(bufDec, bufDecT, P(D_LN2G), P(D_LN2B), bufDecS, (int)ND);
  decomp(bufDecS, bufDec, BB, TDQ);

  gemm(bufDec, P(D_FF1W), P(D_FF1B), bufFF, (int)ND, FF, DD);
  k_gelu<<<cdiv((long long)(ND * FF), 256), 256, 0, stream>>>(bufFF, ND * FF);
  gemm(bufFF, P(D_FF2W), P(D_FF2B), bufDecT, (int)ND, DD, FF);
  addln(bufDec, bufDecT, P(D_LN3G), P(D_LN3B), bufDecS, (int)ND);
  decomp(bufDecS, bufDec, BB, TDQ);

  addln(nullptr, bufDec, P(DEC_NORM_G), P(DEC_NORM_B), bufDecT, (int)ND);
  k_pool_tail<<<cdiv(BB * DD, 256), 256, 0, stream>>>(bufDecT, pooled, BB, TDQ, DD, PREDN);

  float* out_mean = (float*)d_out;
  float* out_lv   = (float*)d_out + BB * PREDN;
  gemm(pooled, P(MEAN_W), P(MEAN_B), out_mean, BB, PREDN, DD);
  gemm(pooled, P(LV_W),   P(LV_B),   out_lv,   BB, PREDN, DD);
  k_clip<<<cdiv(BB * PREDN, 256), 256, 0, stream>>>(out_lv, BB * PREDN, -10.f, 10.f);
}
